// GatedDeltaNetBlock_9088150798894
// MI455X (gfx1250) — matrix operand order repaired and checked by reading, not compiled, not run on hardware
//
#include <hip/hip_runtime.h>
#include <math.h>

// Problem constants (reference: B=4, S=4096, H=2048)
#define HDIM 2048
#define BDIM 4
#define SDIM 4096
#define MTOK (BDIM * SDIM)   // 16384 tokens
#define NCHAN (BDIM * HDIM)  // 8192 scan channels
#define NCHUNK 64
#define CLEN (SDIM / NCHUNK) // 64 steps per chunk
#define KSTEP 32
#define LDSPITCH 40          // 32 bf16 + 8 bf16 pad per row

typedef __bf16 v16bf __attribute__((ext_vector_type(16)));
typedef float  v8f   __attribute__((ext_vector_type(8)));
typedef unsigned int u32x4 __attribute__((ext_vector_type(4)));
typedef int i32x4 __attribute__((ext_vector_type(4)));
typedef int i32x8 __attribute__((ext_vector_type(8)));

__device__ __forceinline__ unsigned short f2bf(float f) {
  union { float f; unsigned u; } c; c.f = f;
  unsigned u = c.u;
  return (unsigned short)((u + 0x7FFFu + ((u >> 16) & 1u)) >> 16); // RNE
}
__device__ __forceinline__ float bf2f(unsigned short s) {
  union { unsigned u; float f; } c; c.u = ((unsigned)s) << 16;
  return c.f;
}

// ---------------------------------------------------------------- TDM
// Issue a Tensor Data Mover load of a 2D bf16 tile (tileW x tileH, row stride
// rowStride elements) into LDS at byte offset ldsOff, with LDS padding of
// 4 DWORDs (8 bf16) after every 16 DWORDs (one 32-elem bf16 row) -> the LDS
// pitch is LDSPITCH elements, matching the fragment-gather addressing.
// D# encoding per CDNA5 ISA 8.3/8.4 (group0: flags/lds/global, group1: dims).
__device__ __forceinline__ void tdm_load_2d(unsigned ldsOff, const void* gaddr,
                                            unsigned tensorW, unsigned tensorH,
                                            unsigned rowStride, unsigned tileW,
                                            unsigned tileH) {
  unsigned long long ga = (unsigned long long)(size_t)gaddr;
  u32x4 g0;
  g0[0] = 1u;                                   // count=1, user descriptor
  g0[1] = ldsOff;                               // lds_addr (bytes)
  g0[2] = (unsigned)(ga & 0xFFFFFFFFu);         // global_addr[31:0]
  g0[3] = (unsigned)((ga >> 32) & 0x1FFFFFFu)   // global_addr[56:32]
        | (2u << 30);                           // type = 2 ("image")
  i32x8 g1;
  g1[0] = (1 << 16)                             // data_size = 2 bytes
        | (1 << 20)                             // pad_enable
        | (3 << 22)                             // pad_interval = 16 DWORDs
        | (3 << 25);                            // pad_amount   = 4 DWORDs
  g1[1] = (int)((tensorW & 0xFFFFu) << 16);     // tensor_dim0[15:0]
  g1[2] = (int)(((tensorW >> 16) & 0xFFFFu) | ((tensorH & 0xFFFFu) << 16));
  g1[3] = (int)(((tensorH >> 16) & 0xFFFFu) | ((tileW & 0xFFFFu) << 16));
  g1[4] = (int)(tileH & 0xFFFFu);               // tile_dim1 (tile_dim2 = 0)
  g1[5] = (int)rowStride;                       // tensor_dim0_stride[31:0]
  g1[6] = 0;
  g1[7] = 0;
  i32x4 g2 = {0, 0, 0, 0};
  i32x4 g3 = {0, 0, 0, 0};
#if defined(__clang_major__) && __clang_major__ >= 23
  i32x8 g4 = {0, 0, 0, 0, 0, 0, 0, 0};
  __builtin_amdgcn_tensor_load_to_lds(g0, g1, g2, g3, g4, 0);
#else
  __builtin_amdgcn_tensor_load_to_lds(g0, g1, g2, g3, 0);
#endif
}

// ---------------------------------------------------------------- converts
__global__ void cvt_f32_bf16(const float* __restrict__ in,
                             unsigned short* __restrict__ out, int n) {
  int stride = gridDim.x * blockDim.x;
  for (int i = blockIdx.x * blockDim.x + threadIdx.x; i < n; i += stride)
    out[i] = f2bf(in[i]);
}

// ---------------------------------------------------------------- WMMA GEMM
// C[m,n] = act( sum_k A[m,k] * W[n,k] (+ bias[n]) )
// 256 threads = 8 waves; block tile 128(M) x 128(N); K-step 32.
// Waves: wm = wave&3 -> 32-row M slice, wn = wave>>2 -> 64-col N slice.
// Each wave: 2x4 accumulators of 16x16 f32 (v8f each).
// Global->LDS staging is done by the Tensor Data Mover (wave 0 issues one
// descriptor per tile), double-buffered against the WMMA compute.
// act: 0 = tanh, 1 = sigmoid(+bias), 2 = identity.
template <bool F32OUT>
__global__ __launch_bounds__(256) void wmma_gemm(
    const unsigned short* __restrict__ Abf,
    const unsigned short* __restrict__ Wbf,
    const float* __restrict__ bias, int actType, void* __restrict__ Cout) {
  __shared__ __align__(16) unsigned short Al[2][128 * LDSPITCH];
  __shared__ __align__(16) unsigned short Bl[2][128 * LDSPITCH];

  const int m0   = blockIdx.x * 128;
  const int n0   = blockIdx.y * 128;
  const int tid  = threadIdx.x;
  const int lane = tid & 31;
  const int wave = tid >> 5;
  const int wm   = wave & 3;   // M slice (32 rows)
  const int wn   = wave >> 2;  // N slice (64 cols)

  v8f acc[2][4] = {};

  // fragment gather coords (ISA 7.12.2 bf16 layouts, wave32)
  const int aRow = wm * 32 + (lane & 15);  // + {0,16} for the two sub-tiles
  const int aK   = (lane < 16) ? 0 : 8;    // A lane K runs {aK..aK+7, aK+16..aK+23}
  const int bN   = lane & 15;
  const int bK   = (lane < 16) ? 0 : 8;    // B lane K runs {bK..bK+7, bK+16..bK+23}

  const unsigned short* aTile = Abf + (size_t)m0 * HDIM;
  const unsigned short* wTile = Wbf + (size_t)n0 * HDIM;
  const unsigned ldsA0 = (unsigned)(size_t)(void*)&Al[0][0];
  const unsigned ldsA1 = (unsigned)(size_t)(void*)&Al[1][0];
  const unsigned ldsB0 = (unsigned)(size_t)(void*)&Bl[0][0];
  const unsigned ldsB1 = (unsigned)(size_t)(void*)&Bl[1][0];

  if (wave == 0) {  // prime buffer 0
    tdm_load_2d(ldsA0, aTile, HDIM, MTOK, HDIM, KSTEP, 128);
    tdm_load_2d(ldsB0, wTile, HDIM, HDIM, HDIM, KSTEP, 128);
  }

  const int nIter = HDIM / KSTEP;
  for (int i = 0; i < nIter; ++i) {
    const int cur = i & 1;
    if (wave == 0) {
      if (i + 1 < nIter) {  // kick DMA for next K-panel into the other buffer
        const int k1 = (i + 1) * KSTEP;
        tdm_load_2d(cur ? ldsA0 : ldsA1, aTile + k1, HDIM, MTOK, HDIM, KSTEP, 128);
        tdm_load_2d(cur ? ldsB0 : ldsB1, wTile + k1, HDIM, HDIM, HDIM, KSTEP, 128);
        __builtin_amdgcn_s_wait_tensorcnt(2);  // current buffer's pair done
      } else {
        __builtin_amdgcn_s_wait_tensorcnt(0);
      }
    }
    __syncthreads();  // publish TDM-written LDS to all waves

    union { uint4 q[2]; v16bf v; } fa[2], fb;
    fa[0].q[0] = *(const uint4*)&Al[cur][(aRow)      * LDSPITCH + aK];
    fa[0].q[1] = *(const uint4*)&Al[cur][(aRow)      * LDSPITCH + aK + 16];
    fa[1].q[0] = *(const uint4*)&Al[cur][(aRow + 16) * LDSPITCH + aK];
    fa[1].q[1] = *(const uint4*)&Al[cur][(aRow + 16) * LDSPITCH + aK + 16];
#pragma unroll
    for (int t = 0; t < 4; ++t) {
      const int brow = wn * 64 + t * 16 + bN;
      fb.q[0] = *(const uint4*)&Bl[cur][brow * LDSPITCH + bK];
      fb.q[1] = *(const uint4*)&Bl[cur][brow * LDSPITCH + bK + 16];
      acc[0][t] = __builtin_amdgcn_wmma_f32_16x16x32_bf16(
          false, fa[0].v, false, fb.v, (short)0, acc[0][t], false, false);
      acc[1][t] = __builtin_amdgcn_wmma_f32_16x16x32_bf16(
          false, fa[1].v, false, fb.v, (short)0, acc[1][t], false, false);
    }
    __syncthreads();  // all waves done reading before buffer is re-filled
  }

  // epilogue: C 16x16 f32 layout — VGPR r: lanes0-15 -> M=r, lanes16-31 -> M=8+r
  const int erow = (lane < 16) ? 0 : 8;
  const int ecol = lane & 15;
#pragma unroll
  for (int s = 0; s < 2; ++s) {
#pragma unroll
    for (int t = 0; t < 4; ++t) {
      const int n = n0 + wn * 64 + t * 16 + ecol;
      const float bi = (actType == 1) ? bias[n] : 0.f;
#pragma unroll
      for (int r = 0; r < 8; ++r) {
        const int m = m0 + wm * 32 + s * 16 + erow + r;
        float v = acc[s][t][r];
        if (actType == 0)      v = tanhf(v);
        else if (actType == 1) v = 1.f / (1.f + __expf(-(v + bi)));
        if constexpr (F32OUT)
          ((float*)Cout)[(size_t)m * HDIM + n] = v;
        else
          ((unsigned short*)Cout)[(size_t)m * HDIM + n] = f2bf(v);
      }
    }
  }
}

// ---------------------------------------------------------------- scan
// st' = m ? (a*st + beta*(v - st*k)) : st  ==  A_t*st + B_t  (linear, associative)
// A_t = m ? (a - beta*k) : 1 ;  B_t = m ? beta*v : 0

__global__ void scan_pass1(const unsigned short* __restrict__ kb,
                           const unsigned short* __restrict__ vb,
                           const unsigned short* __restrict__ ab,
                           const unsigned short* __restrict__ btb,
                           const unsigned char* __restrict__ mask,
                           float* __restrict__ chunkA,
                           float* __restrict__ chunkB) {
  const int idx = blockIdx.x * blockDim.x + threadIdx.x; // chunk*NCHAN + b*HDIM + h
  if (idx >= NCHAN * NCHUNK) return;
  const int h = idx % HDIM;
  const int b = (idx / HDIM) % BDIM;
  const int chunk = idx / NCHAN;
  float Ap = 1.f, Bp = 0.f;
  const int t0 = chunk * CLEN;
  for (int i = 0; i < CLEN; ++i) {
    const int t = t0 + i;
    const size_t off = ((size_t)(b * SDIM + t)) * HDIM + h;
    const bool m = mask[b * SDIM + t] != 0;
    const float kk = bf2f(kb[off]), vv = bf2f(vb[off]);
    const float aa = bf2f(ab[off]), be = bf2f(btb[off]);
    const float At = m ? (aa - be * kk) : 1.f;
    const float Bt = m ? (be * vv) : 0.f;
    Ap = At * Ap;
    Bp = At * Bp + Bt;
  }
  chunkA[idx] = Ap;
  chunkB[idx] = Bp;
}

__global__ void scan_pass2(const float* __restrict__ state,
                           const float* __restrict__ chunkA,
                           const float* __restrict__ chunkB,
                           float* __restrict__ stStart,
                           float* __restrict__ finalState) {
  const int ch = blockIdx.x * blockDim.x + threadIdx.x;
  if (ch >= NCHAN) return;
  float st = state[ch];
  for (int c = 0; c < NCHUNK; ++c) {
    stStart[c * NCHAN + ch] = st;
    st = chunkA[c * NCHAN + ch] * st + chunkB[c * NCHAN + ch];
  }
  finalState[ch] = st;
}

__global__ void scan_pass3(const unsigned short* __restrict__ kb,
                           const unsigned short* __restrict__ vb,
                           const unsigned short* __restrict__ ab,
                           const unsigned short* __restrict__ btb,
                           const unsigned short* __restrict__ qb,
                           const unsigned char* __restrict__ mask,
                           const float* __restrict__ stStart,
                           unsigned short* __restrict__ yb) {
  const int idx = blockIdx.x * blockDim.x + threadIdx.x;
  if (idx >= NCHAN * NCHUNK) return;
  const int h = idx % HDIM;
  const int b = (idx / HDIM) % BDIM;
  const int chunk = idx / NCHAN;
  float st = stStart[idx];
  const int t0 = chunk * CLEN;
  for (int i = 0; i < CLEN; ++i) {
    const int t = t0 + i;
    const size_t off = ((size_t)(b * SDIM + t)) * HDIM + h;
    const bool m = mask[b * SDIM + t] != 0;
    const float kk = bf2f(kb[off]), vv = bf2f(vb[off]);
    const float aa = bf2f(ab[off]), be = bf2f(btb[off]);
    if (m) st = aa * st + be * (vv - st * kk);
    const float out = m ? bf2f(qb[off]) * st : 0.f;
    yb[off] = f2bf(out);
  }
}

// ---------------------------------------------------------------- launch
extern "C" void kernel_launch(void* const* d_in, const int* in_sizes, int n_in,
                              void* d_out, int out_size, void* d_ws,
                              size_t ws_size, hipStream_t stream) {
  const float* x     = (const float*)d_in[0];
  const float* state = (const float*)d_in[1];
  const unsigned char* mask = (const unsigned char*)d_in[2];
  const float* Wq = (const float*)d_in[3];
  const float* Wk = (const float*)d_in[4];
  const float* Wv = (const float*)d_in[5];
  const float* Wa = (const float*)d_in[6];
  const float* ba = (const float*)d_in[7];
  const float* Wb = (const float*)d_in[8];
  const float* bb = (const float*)d_in[9];
  const float* Wo = (const float*)d_in[10];

  float* y_out       = (float*)d_out;
  float* final_state = (float*)d_out + (size_t)MTOK * HDIM;

  // workspace bump allocator (256B aligned)
  char* ws = (char*)d_ws;
  auto alloc = [&](size_t bytes) -> void* {
    void* p = ws;
    ws += (bytes + 255) & ~(size_t)255;
    return p;
  };
  const size_t actBytes = (size_t)MTOK * HDIM * sizeof(unsigned short);
  const size_t wBytes   = (size_t)HDIM * HDIM * sizeof(unsigned short);
  unsigned short* xb  = (unsigned short*)alloc(actBytes);
  unsigned short* wqb = (unsigned short*)alloc(wBytes);
  unsigned short* wkb = (unsigned short*)alloc(wBytes);
  unsigned short* wvb = (unsigned short*)alloc(wBytes);
  unsigned short* wab = (unsigned short*)alloc(wBytes);
  unsigned short* wbb = (unsigned short*)alloc(wBytes);
  unsigned short* wob = (unsigned short*)alloc(wBytes);
  unsigned short* aq  = (unsigned short*)alloc(actBytes);
  unsigned short* ak  = (unsigned short*)alloc(actBytes);
  unsigned short* av  = (unsigned short*)alloc(actBytes);
  unsigned short* aa  = (unsigned short*)alloc(actBytes);
  unsigned short* ab  = (unsigned short*)alloc(actBytes);
  unsigned short* yb  = (unsigned short*)alloc(actBytes);
  float* chunkA  = (float*)alloc((size_t)NCHAN * NCHUNK * sizeof(float));
  float* chunkB  = (float*)alloc((size_t)NCHAN * NCHUNK * sizeof(float));
  float* stStart = (float*)alloc((size_t)NCHAN * NCHUNK * sizeof(float));

  // 1) fp32 -> bf16 demotion (x stays resident in the 192MB L2 for all 5 projs)
  cvt_f32_bf16<<<2048, 256, 0, stream>>>(x, xb, MTOK * HDIM);
  cvt_f32_bf16<<<1024, 256, 0, stream>>>(Wq, wqb, HDIM * HDIM);
  cvt_f32_bf16<<<1024, 256, 0, stream>>>(Wk, wkb, HDIM * HDIM);
  cvt_f32_bf16<<<1024, 256, 0, stream>>>(Wv, wvb, HDIM * HDIM);
  cvt_f32_bf16<<<1024, 256, 0, stream>>>(Wa, wab, HDIM * HDIM);
  cvt_f32_bf16<<<1024, 256, 0, stream>>>(Wb, wbb, HDIM * HDIM);
  cvt_f32_bf16<<<1024, 256, 0, stream>>>(Wo, wob, HDIM * HDIM);

  // 2) five projection GEMMs with fused activations (bf16 out)
  dim3 gemmGrid(MTOK / 128, HDIM / 128);
  wmma_gemm<false><<<gemmGrid, 256, 0, stream>>>(xb, wqb, nullptr, 0, aq);
  wmma_gemm<false><<<gemmGrid, 256, 0, stream>>>(xb, wkb, nullptr, 0, ak);
  wmma_gemm<false><<<gemmGrid, 256, 0, stream>>>(xb, wvb, nullptr, 2, av);
  wmma_gemm<false><<<gemmGrid, 256, 0, stream>>>(xb, wab, ba, 1, aa);
  wmma_gemm<false><<<gemmGrid, 256, 0, stream>>>(xb, wbb, bb, 1, ab);

  // 3) associative linear scan: chunk reduce -> chunk combine -> replay
  scan_pass1<<<(NCHAN * NCHUNK) / 256, 256, 0, stream>>>(ak, av, aa, ab, mask,
                                                         chunkA, chunkB);
  scan_pass2<<<NCHAN / 256, 256, 0, stream>>>(state, chunkA, chunkB, stStart,
                                              final_state);
  scan_pass3<<<(NCHAN * NCHUNK) / 256, 256, 0, stream>>>(ak, av, aa, ab, aq,
                                                         mask, stStart, yb);

  // 4) output GEMM (f32 out, straight into d_out)
  wmma_gemm<true><<<gemmGrid, 256, 0, stream>>>(yb, wob, nullptr, 2, y_out);
}
